// SNN2_21878563406060
// MI455X (gfx1250) — matrix operand order repaired and checked by reading, not compiled, not run on hardware
//
#include <hip/hip_runtime.h>

// ---------------------------------------------------------------------------
// SNN forward on gfx1250 (MI455X).
//   x  : [2048, 50, 620] f32
//   W1 : [620, 620] f32   (layer1, no bias)
//   W2 : [11, 620]  f32   (layer2, no bias)
//   out: [2048, 11] f32   (mean spike count over T=50)
//
// Batch-parallel blocks (M=32 rows) carry the full T=50 recurrence.
// Layer-1 GEMM runs on v_wmma_f32_16x16x32_f16 with membrane state held in
// the WMMA C/D fragment layout in VGPRs. x_{t+1} is prefetched with
// GLOBAL_LOAD_ASYNC_TO_LDS_B128 (ASYNCcnt) while step t computes; the f32->f16
// conversion is a shared LDS->LDS pass. Spikes go to LDS as f16 {0,1} and feed
// the layer-2 WMMA. Weights are pre-converted to zero-padded f16.
// ---------------------------------------------------------------------------

typedef __attribute__((ext_vector_type(16))) _Float16 v16h;
typedef __attribute__((ext_vector_type(8)))  _Float16 v8h;
typedef __attribute__((ext_vector_type(4)))  _Float16 v4h;
typedef __attribute__((ext_vector_type(8)))  float    v8f;
typedef __attribute__((ext_vector_type(4)))  float    v4f;

#define B_TOT   2048
#define T_STEPS 50
#define N_IN    620
#define N_H     620
#define N_CLS   11
#define KP      640      // padded K / N (20 WMMA K-chunks of 32)
#define MT      32       // batch rows per block (2 WMMA M-tiles)
#define SAPAD   648      // sA row stride in halves (1296 B: 16B-aligned, bank-spread)
#define SSPAD   640      // sS row stride in halves (1280 B: 16B-aligned)
#define NQ      (KP / 4) // 160 quads per row

// A-fragment (16x32 f16, MxK): per lane, halves 0-15 = K(+0..15) contiguous,
// lane>=16 half-wave shifted +16 in K. Two 16B LDS loads.
__device__ __forceinline__ v16h frag_a(const _Float16* p) {
    v8h lo = *(const v8h*)p;
    v8h hi = *(const v8h*)(p + 8);
    return __builtin_shufflevector(lo, hi, 0,1,2,3,4,5,6,7,8,9,10,11,12,13,14,15);
}

// B-fragment (32x16 f16, KxN): per lane, 16 contiguous K-halves of column (lane&15),
// lanes>=16 hold K+16..31. One aligned 32B global load.
__device__ __forceinline__ v16h frag_b(const _Float16* p) {
    return *(const v16h*)p;
}

// --- weight conversion pre-passes (f32 -> zero-padded f16, [n][k] layout) ---
__global__ void cvt_w1(const float* __restrict__ W1, _Float16* __restrict__ W1h) {
    int i = blockIdx.x * 256 + threadIdx.x;          // KP*KP threads
    int n = i / KP, k = i - n * KP;
    W1h[i] = (n < N_H && k < N_IN) ? (_Float16)W1[n * N_IN + k] : (_Float16)0.0f;
}
__global__ void cvt_w2(const float* __restrict__ W2, _Float16* __restrict__ W2h) {
    int i = blockIdx.x * 256 + threadIdx.x;          // 16*KP threads
    int n = i / KP, k = i - n * KP;
    W2h[i] = (n < N_CLS && k < N_H) ? (_Float16)W2[n * N_H + k] : (_Float16)0.0f;
}

// --- main recurrent kernel ---
__global__ __launch_bounds__(256) void snn_step_kernel(
        const float*    __restrict__ x,
        const _Float16* __restrict__ W1h,
        const _Float16* __restrict__ W2h,
        float*          __restrict__ out) {

    __shared__ float    sX[MT * N_IN];    // raw f32 x_t staging (async-copy target)
    __shared__ _Float16 sA[MT * SAPAD];   // f16 x_t tile     [32][648]
    __shared__ _Float16 sS[MT * SSPAD];   // f16 h1 spikes    [32][640]

    const int tid   = threadIdx.x;
    const int wave  = tid >> 5;
    const int lane  = tid & 31;
    const int b0    = blockIdx.x * MT;
    const int rlo   = lane & 15;              // N (or M) index within a 16-tile
    const int khalf = (lane >> 4) << 4;       // A-fragment K sub-offset: 0 or 16
    const int kbq   = (lane >> 4) << 4;       // B-fragment K sub-offset: 0 or 16
    const int rowb  = (lane >> 4) << 3;       // C/D row sub-offset: 0 or 8

    // persistent layer-1 membrane state, WMMA C/D layout: [m][j][r]
    float st[80];
#pragma unroll
    for (int i = 0; i < 80; ++i) st[i] = 0.0f;

    // layer-2 state + spike accumulator (only waves 0,1 use these)
    float h2m[8], sums[8];
#pragma unroll
    for (int r = 0; r < 8; ++r) { h2m[r] = 0.0f; sums[r] = 0.0f; }

    // ---- prologue: async-stage x_0 (each thread posts its own quads) ----
    for (int idx = tid; idx < MT * NQ; idx += 256) {
        int row = idx / NQ;
        int k   = (idx - row * NQ) << 2;
        if (k < N_IN) {
            const float* g = x + ((size_t)(b0 + row) * T_STEPS + 0) * N_IN + k;
            unsigned lds = (unsigned)(uintptr_t)(&sX[row * N_IN + k]);
            asm volatile("global_load_async_to_lds_b128 %0, %1, off"
                         :: "v"(lds), "v"(g) : "memory");
        }
    }

    for (int t = 0; t < T_STEPS; ++t) {
        // ---- wait for this thread's async staging of x_t, then convert to f16 ----
        asm volatile("s_wait_asynccnt 0x0" ::: "memory");
        for (int idx = tid; idx < MT * NQ; idx += 256) {
            int row = idx / NQ;
            int k   = (idx - row * NQ) << 2;
            v4h h;
            h.x = (_Float16)0.0f; h.y = (_Float16)0.0f;
            h.z = (_Float16)0.0f; h.w = (_Float16)0.0f;
            if (k < N_IN) {   // quads 155..159 stay zero pad
                v4f v = *(const v4f*)(&sX[row * N_IN + k]);
                h.x = (_Float16)v.x; h.y = (_Float16)v.y;
                h.z = (_Float16)v.z; h.w = (_Float16)v.w;
            }
            *(v4h*)(&sA[row * SAPAD + k]) = h;
        }
        // ---- post async staging of x_{t+1}; this thread's sX reads are done ----
        if (t + 1 < T_STEPS) {
            for (int idx = tid; idx < MT * NQ; idx += 256) {
                int row = idx / NQ;
                int k   = (idx - row * NQ) << 2;
                if (k < N_IN) {
                    const float* g = x + ((size_t)(b0 + row) * T_STEPS + (t + 1)) * N_IN + k;
                    unsigned lds = (unsigned)(uintptr_t)(&sX[row * N_IN + k]);
                    asm volatile("global_load_async_to_lds_b128 %0, %1, off"
                                 :: "v"(lds), "v"(g) : "memory");
                }
            }
        }
        __syncthreads();

        // ---- layer 1: [32 x 640] = A(32x640) x W1h^T, per-wave 2x5 output tiles ----
        v8f acc0[5], acc1[5];
#pragma unroll
        for (int j = 0; j < 5; ++j) {
            acc0[j] = (v8f){0,0,0,0,0,0,0,0};
            acc1[j] = (v8f){0,0,0,0,0,0,0,0};
        }
#pragma unroll 1
        for (int kc = 0; kc < 20; ++kc) {
            const _Float16* pa = &sA[rlo * SAPAD + kc * 32 + khalf];
            v16h a0 = frag_a(pa);
            v16h a1 = frag_a(pa + 16 * SAPAD);
#pragma unroll
            for (int j = 0; j < 5; ++j) {
                const _Float16* pb =
                    W1h + (size_t)((wave * 5 + j) * 16 + rlo) * KP + kc * 32 + kbq;
                v16h b = frag_b(pb);
                acc0[j] = __builtin_amdgcn_wmma_f32_16x16x32_f16(
                              false, a0, false, b, (short)0, acc0[j], false, false);
                acc1[j] = __builtin_amdgcn_wmma_f32_16x16x32_f16(
                              false, a1, false, b, (short)0, acc1[j], false, false);
            }
        }

        // ---- layer-1 membrane update + spike emit (elementwise on fragments) ----
#pragma unroll
        for (int j = 0; j < 5; ++j) {
            const int col = (wave * 5 + j) * 16 + rlo;
#pragma unroll
            for (int m = 0; m < 2; ++m) {
#pragma unroll
                for (int r = 0; r < 8; ++r) {
                    float s = st[(m * 5 + j) * 8 + r];
                    s = (s > 0.5f) ? 0.0f : s * 0.2f;        // leak, reset if spiked
                    float g = m ? acc1[j][r] : acc0[j][r];
                    s += (g > 0.0f) ? g : 0.0f;              // + relu(x W1^T)
                    st[(m * 5 + j) * 8 + r] = s;
                    sS[(m * 16 + rowb + r) * SSPAD + col] =
                        (s > 0.5f) ? (_Float16)1.0f : (_Float16)0.0f;
                }
            }
        }
        __syncthreads();

        // ---- layer 2: spikes(32x640) x W2h^T(640x16-pad), waves 0..1 only ----
        if (wave < 2) {
            v8f c2 = (v8f){0,0,0,0,0,0,0,0};
#pragma unroll 1
            for (int kc = 0; kc < 20; ++kc) {
                const _Float16* pa = &sS[(wave * 16 + rlo) * SSPAD + kc * 32 + khalf];
                v16h a = frag_a(pa);
                const _Float16* pb = W2h + (size_t)rlo * KP + kc * 32 + kbq;
                v16h b = frag_b(pb);
                c2 = __builtin_amdgcn_wmma_f32_16x16x32_f16(
                         false, a, false, b, (short)0, c2, false, false);
            }
#pragma unroll
            for (int r = 0; r < 8; ++r) {
                float h = h2m[r];
                h = (h > 0.5f) ? 0.0f : h * 0.2f;
                float g = c2[r];
                h += (g > 0.0f) ? g : 0.0f;
                h2m[r] = h;
                sums[r] += (h > 0.5f) ? 1.0f : 0.0f;
            }
        }
        // next iteration's barrier orders sS reuse vs. layer-2 reads; sX and sA
        // are rewritten only after all waves passed this iteration's sync points.
    }

    // ---- write output: out[b, c] = sumspike / 50 ----
    if (wave < 2 && rlo < N_CLS) {
#pragma unroll
        for (int r = 0; r < 8; ++r) {
            out[(size_t)(b0 + wave * 16 + rowb + r) * N_CLS + rlo] = sums[r] * 0.02f;
        }
    }
}

extern "C" void kernel_launch(void* const* d_in, const int* in_sizes, int n_in,
                              void* d_out, int out_size, void* d_ws, size_t ws_size,
                              hipStream_t stream) {
    (void)in_sizes; (void)n_in; (void)out_size; (void)ws_size;
    const float* x  = (const float*)d_in[0];
    const float* W1 = (const float*)d_in[1];
    const float* W2 = (const float*)d_in[2];

    _Float16* W1h = (_Float16*)d_ws;                                   // 640*640 f16
    _Float16* W2h = (_Float16*)((char*)d_ws + (size_t)KP * KP * 2);    //  16*640 f16

    cvt_w1<<<(KP * KP) / 256, 256, 0, stream>>>(W1, W1h);
    cvt_w2<<<(16 * KP) / 256, 256, 0, stream>>>(W2, W2h);
    snn_step_kernel<<<B_TOT / MT, 256, 0, stream>>>(x, W1h, W2h, (float*)d_out);
}
